// MultiHeadAttention_11123965297244
// MI455X (gfx1250) — matrix operand order repaired and checked by reading, not compiled, not run on hardware
//
#include <hip/hip_runtime.h>

// ---------------------------------------------------------------------------
// MHA for MI455X (gfx1250): bf16 WMMA everywhere, fp32 accumulate.
// Pipeline: 3x projection GEMM -> flash attention -> output GEMM.
// ---------------------------------------------------------------------------

typedef __bf16 bf16;
typedef __attribute__((ext_vector_type(16))) __bf16 v16bf;
typedef __attribute__((ext_vector_type(8)))  float  v8f;

#define D_MODEL 1024
#define N_HEADS 16
#define HEAD_DIM 64
#define BATCH 2
#define SEQ 2048
#define M_TOTAL (BATCH * SEQ)   // 4096 rows

static __device__ __forceinline__ bf16 f2bf(float f) { return (bf16)f; }

// ---------------------------------------------------------------------------
// GEMM: Y = X @ W^T + bias.  X:[4096,1024] (fp32 if QKV, bf16 if !QKV),
// W:[1024,1024] fp32, bias:[1024] fp32.
// QKV==1: Y scattered bf16 into [B,H,S,hd] head-major layout.
// QKV==0: Y fp32 row-major [4096,1024] (final output).
// Block tile 128(M) x 64(N) x 32(K); 256 threads = 8 waves; per wave:
// one 16-row A fragment x 4 WMMA accumulators (16x16 each).
// ---------------------------------------------------------------------------
template <int QKV>
__global__ __launch_bounds__(256) void gemm_wmma_kernel(const void* __restrict__ Xv,
                                                        const float* __restrict__ W,
                                                        const float* __restrict__ bias,
                                                        void* __restrict__ Yv) {
  constexpr int MT = 128, NT = 64, KT = 32;
  constexpr int LDP = KT + 8;  // padded LDS stride (bank-conflict avoidance)
  __shared__ bf16 sA[MT][LDP];
  __shared__ bf16 sB[NT][LDP];

  const int tid  = threadIdx.x;
  const int wave = tid >> 5;
  const int lane = tid & 31;
  const int hi   = lane >> 4;   // 0: lanes 0-15, 1: lanes 16-31
  const int lo16 = lane & 15;

  const int gm0 = blockIdx.x * MT;   // 32 blocks
  const int gn0 = blockIdx.y * NT;   // 16 blocks

  const float* Xf = (const float*)Xv;
  const bf16*  Xh = (const bf16*)Xv;

  v8f acc[4];
#pragma unroll
  for (int nt = 0; nt < 4; ++nt)
#pragma unroll
    for (int j = 0; j < 8; ++j) acc[nt][j] = 0.0f;

  for (int k0 = 0; k0 < D_MODEL; k0 += KT) {
    __syncthreads();
    // Stage A tile (128x32) fp32/bf16 -> bf16 LDS. 2 threads/row, 16 elems each.
    {
      const int row = tid >> 1;
      const int seg = (tid & 1) * 16;
      const size_t base = (size_t)(gm0 + row) * D_MODEL + k0 + seg;
#pragma unroll
      for (int i = 0; i < 16; ++i) {
        float v;
        if constexpr (QKV) v = Xf[base + i];
        else               v = (float)Xh[base + i];
        sA[row][seg + i] = f2bf(v);
      }
    }
    // Stage B tile: rows are W rows (N dim), contiguous in K. 4 threads/row.
    {
      const int row = tid >> 2;
      const int seg = (tid & 3) * 8;
      const size_t base = (size_t)(gn0 + row) * D_MODEL + k0 + seg;
#pragma unroll
      for (int i = 0; i < 8; ++i) sB[row][seg + i] = f2bf(W[base + i]);
    }
    __syncthreads();

    // A fragment (16x32 bf16): lane<16 row=lane K{0-7,16-23}; lane>=16 row=lane-16 K{8-15,24-31}
    v16bf a;
    {
      const int row = wave * 16 + lo16;
#pragma unroll
      for (int i = 0; i < 8; ++i) a[i]     = sA[row][hi * 8 + i];
#pragma unroll
      for (int i = 0; i < 8; ++i) a[8 + i] = sA[row][16 + hi * 8 + i];
    }
#pragma unroll
    for (int nt = 0; nt < 4; ++nt) {
      // B fragment (32x16 bf16): column = lo16; lanes 0-15 hold K{0-7,16-23}, lanes 16-31 K{8-15,24-31}
      v16bf bfr;
      const int nrow = nt * 16 + lo16;
#pragma unroll
      for (int i = 0; i < 8; ++i) bfr[i] = sB[nrow][hi * 8 + i];
#pragma unroll
      for (int i = 0; i < 8; ++i) bfr[8 + i] = sB[nrow][16 + hi * 8 + i];
      acc[nt] = __builtin_amdgcn_wmma_f32_16x16x32_bf16(false, a, false, bfr,
                                                        (short)0, acc[nt], false, false);
    }
  }

  // Epilogue. C layout: VGPR j -> row (j + hi*8), col lo16.
#pragma unroll
  for (int nt = 0; nt < 4; ++nt) {
#pragma unroll
    for (int j = 0; j < 8; ++j) {
      const int gm = gm0 + wave * 16 + hi * 8 + j;
      const int gn = gn0 + nt * 16 + lo16;
      const float v = acc[nt][j] + bias[gn];
      if constexpr (QKV) {
        const int h = gn >> 6, dd = gn & 63;          // head, head-dim
        const int b = gm >> 11, s = gm & (SEQ - 1);   // batch, seq
        const size_t idx = (((size_t)(b * N_HEADS + h) * SEQ + s) * HEAD_DIM) + dd;
        ((bf16*)Yv)[idx] = f2bf(v);
      } else {
        ((float*)Yv)[(size_t)gm * D_MODEL + gn] = v;
      }
    }
  }
}

// ---------------------------------------------------------------------------
// Flash attention: one block = (b, h, 64 query rows); 4 waves x 16 q-rows.
// score = mask ? (q.k)/64 : -1e30 ; online softmax; O = softmax(S) @ V.
// Output ws stored bf16 in [B,H,S,hd] (contiguous == torch's buggy reshape).
// ---------------------------------------------------------------------------
__global__ __launch_bounds__(128) void attn_kernel(const bf16* __restrict__ Qh,
                                                   const bf16* __restrict__ Kh,
                                                   const bf16* __restrict__ Vh,
                                                   const int* __restrict__ mask,
                                                   bf16* __restrict__ ws) {
  constexpr int QB = 64;   // query rows per block
  constexpr int KB = 32;   // keys per iteration (one 16x32 A fragment for P@V)
  constexpr int LDP = KB + 8;
  __shared__ bf16 sVt[HEAD_DIM][LDP];   // V^T tile: [dim][key]
  __shared__ bf16 sP[4][16][LDP];       // per-wave P scratch (C-layout -> A-layout)

  const int tid  = threadIdx.x;
  const int wave = tid >> 5;
  const int lane = tid & 31;
  const int hi   = lane >> 4;
  const int lo16 = lane & 15;

  const int blk = blockIdx.x;           // B*H*(S/QB) = 1024
  const int qb  = blk & 31;
  const int h   = (blk >> 5) & (N_HEADS - 1);
  const int b   = blk >> 9;

  const size_t headbase = (size_t)(b * N_HEADS + h) * SEQ * HEAD_DIM;
  const int q0 = qb * QB + wave * 16;

  // Q fragments: two 16x32 bf16 A-fragments covering dims [0,32) and [32,64).
  v16bf aq[2];
  {
    const bf16* qrow = Qh + headbase + (size_t)(q0 + lo16) * HEAD_DIM;
#pragma unroll
    for (int c = 0; c < 2; ++c) {
#pragma unroll
      for (int i = 0; i < 8; ++i) aq[c][i]     = qrow[c * 32 + hi * 8 + i];
#pragma unroll
      for (int i = 0; i < 8; ++i) aq[c][8 + i] = qrow[c * 32 + 16 + hi * 8 + i];
    }
  }

  v8f o[4];
#pragma unroll
  for (int nt = 0; nt < 4; ++nt)
#pragma unroll
    for (int j = 0; j < 8; ++j) o[nt][j] = 0.0f;

  float rmax[8], rsum[8];
#pragma unroll
  for (int j = 0; j < 8; ++j) { rmax[j] = -1e30f; rsum[j] = 0.0f; }

  const float scale = 1.0f / (float)HEAD_DIM;
  const int* mbase = mask + (size_t)b * SEQ * SEQ;

  for (int kb = 0; kb < SEQ; kb += KB) {
    __syncthreads();
    // Stage V^T tile: 32 keys x 64 dims -> sVt[dim][key]. 16 elems/thread.
    {
      const int e0  = tid * 16;
      const int key = e0 >> 6;
      const int d0  = e0 & 63;
      const bf16* vp = Vh + headbase + (size_t)(kb + key) * HEAD_DIM + d0;
#pragma unroll
      for (int i = 0; i < 16; ++i) sVt[d0 + i][key] = vp[i];
    }
    __syncthreads();

    // Scores for two 16-key tiles: S = Q(16x64) @ K^T(64x16) per tile.
    v8f s[2];
#pragma unroll
    for (int t = 0; t < 2; ++t) {
      const int kt0 = kb + t * 16;
      v8f z;
#pragma unroll
      for (int j = 0; j < 8; ++j) z[j] = 0.0f;
#pragma unroll
      for (int c = 0; c < 2; ++c) {
        // B fragment = K^T chunk (32 kdims x 16 keys): column=key=lo16,
        // lanes 0-15 kdim c*32+{0-7,16-23}, lanes 16-31 kdim c*32+{8-15,24-31}.
        v16bf bk;
        const bf16* kp = Kh + headbase + (size_t)(kt0 + lo16) * HEAD_DIM + c * 32 + hi * 8;
#pragma unroll
        for (int i = 0; i < 8; ++i) bk[i] = kp[i];
#pragma unroll
        for (int i = 0; i < 8; ++i) bk[8 + i] = kp[16 + i];
        z = __builtin_amdgcn_wmma_f32_16x16x32_bf16(false, aq[c], false, bk,
                                                    (short)0, z, false, false);
      }
      s[t] = z;
    }

    // Mask, scale, online-softmax stats over these 32 keys.
    float alpha[8];
#pragma unroll
    for (int j = 0; j < 8; ++j) {
      const int q = q0 + hi * 8 + j;
      const int* mr = mbase + (size_t)q * SEQ + kb;
      float v0 = s[0][j] * scale;
      float v1 = s[1][j] * scale;
      v0 = mr[lo16]      ? v0 : -1e30f;
      v1 = mr[16 + lo16] ? v1 : -1e30f;
      float m = fmaxf(v0, v1);
#pragma unroll
      for (int d = 1; d < 16; d <<= 1) m = fmaxf(m, __shfl_xor(m, d));
      const float nm = fmaxf(rmax[j], m);
      const float al = __expf(rmax[j] - nm);
      rmax[j] = nm;
      const float p0 = __expf(v0 - nm);
      const float p1 = __expf(v1 - nm);
      float ps = p0 + p1;
#pragma unroll
      for (int d = 1; d < 16; d <<= 1) ps += __shfl_xor(ps, d);
      rsum[j] = rsum[j] * al + ps;
      alpha[j] = al;
      s[0][j] = p0;
      s[1][j] = p1;
    }

    // Convert P (C-layout) -> bf16 A-layout via per-wave LDS scratch.
#pragma unroll
    for (int j = 0; j < 8; ++j) {
      sP[wave][hi * 8 + j][lo16]      = f2bf(s[0][j]);
      sP[wave][hi * 8 + j][16 + lo16] = f2bf(s[1][j]);
    }
    v16bf pa;
    {
#pragma unroll
      for (int i = 0; i < 8; ++i) pa[i]     = sP[wave][lo16][hi * 8 + i];
#pragma unroll
      for (int i = 0; i < 8; ++i) pa[8 + i] = sP[wave][lo16][16 + hi * 8 + i];
    }

    // O = O*alpha + P(16x32) @ V(32x64); 4 WMMA over dim chunks.
#pragma unroll
    for (int nt = 0; nt < 4; ++nt) {
#pragma unroll
      for (int j = 0; j < 8; ++j) o[nt][j] *= alpha[j];
      v16bf bv;
      const int dim = nt * 16 + lo16;
#pragma unroll
      for (int i = 0; i < 8; ++i) bv[i] = sVt[dim][hi * 8 + i];
#pragma unroll
      for (int i = 0; i < 8; ++i) bv[8 + i] = sVt[dim][16 + hi * 8 + i];
      o[nt] = __builtin_amdgcn_wmma_f32_16x16x32_bf16(false, pa, false, bv,
                                                      (short)0, o[nt], false, false);
    }
  }

  // Epilogue: normalize and store bf16 into [B,H,S,hd].
#pragma unroll
  for (int j = 0; j < 8; ++j) {
    const float inv = 1.0f / rsum[j];
    const int q = q0 + hi * 8 + j;
    bf16* orow = ws + headbase + (size_t)q * HEAD_DIM;
#pragma unroll
    for (int nt = 0; nt < 4; ++nt) orow[nt * 16 + lo16] = f2bf(o[nt][j] * inv);
  }
}

// ---------------------------------------------------------------------------
extern "C" void kernel_launch(void* const* d_in, const int* in_sizes, int n_in,
                              void* d_out, int out_size, void* d_ws, size_t ws_size,
                              hipStream_t stream) {
  (void)in_sizes; (void)n_in; (void)out_size; (void)ws_size;

  const float* query = (const float*)d_in[0];
  const float* key   = (const float*)d_in[1];
  const float* value = (const float*)d_in[2];
  const int*   mask  = (const int*)d_in[3];
  const float* Wq = (const float*)d_in[4];
  const float* bq = (const float*)d_in[5];
  const float* Wk = (const float*)d_in[6];
  const float* bk = (const float*)d_in[7];
  const float* Wv = (const float*)d_in[8];
  const float* bv = (const float*)d_in[9];
  const float* Wo = (const float*)d_in[10];
  const float* bo = (const float*)d_in[11];
  float* out = (float*)d_out;

  const size_t headElems = (size_t)BATCH * N_HEADS * SEQ * HEAD_DIM;  // 4,194,304
  bf16* Qh = (bf16*)d_ws;
  bf16* Kh = Qh + headElems;
  bf16* Vh = Kh + headElems;
  bf16* Ws = Vh + headElems;  // attention output, head-major (== torch reshape memory)

  dim3 ggrid(M_TOTAL / 128, D_MODEL / 64);  // (32, 16)
  gemm_wmma_kernel<1><<<ggrid, 256, 0, stream>>>(query, Wq, bq, Qh);
  gemm_wmma_kernel<1><<<ggrid, 256, 0, stream>>>(key,   Wk, bk, Kh);
  gemm_wmma_kernel<1><<<ggrid, 256, 0, stream>>>(value, Wv, bv, Vh);

  attn_kernel<<<BATCH * N_HEADS * (SEQ / 64), 128, 0, stream>>>(Qh, Kh, Vh, mask, Ws);

  gemm_wmma_kernel<0><<<ggrid, 256, 0, stream>>>(Ws, Wo, bo, out);
}
